// MultiHeadDifferentialAttention_13443247637323
// MI455X (gfx1250) — compile-verified, matrix operand order checked by reading, not run on hardware
//
#include <hip/hip_runtime.h>

// Problem constants (fixed by the reference).
#define B_ 2
#define S_ 2048
#define E_ 2048
#define H_ 16
#define D_ 128

typedef __bf16 bf16;
typedef __attribute__((ext_vector_type(4)))  float    v4f;
typedef __attribute__((ext_vector_type(8)))  float    v8f;
typedef __attribute__((ext_vector_type(4)))  __bf16   v4bf;
typedef __attribute__((ext_vector_type(8)))  __bf16   v8bf;
typedef __attribute__((ext_vector_type(16))) __bf16   v16bf;
typedef __attribute__((ext_vector_type(4)))  unsigned v4u;
typedef __attribute__((ext_vector_type(8)))  int      v8i;
typedef __attribute__((ext_vector_type(4)))  int      v4i;

union FragBF {
    v16bf v;
    v8bf  h[2];
};

__device__ __forceinline__ v8f wmma_bf16f32(v16bf a, v16bf b, v8f c) {
    // D = A(16x32 bf16) * B(32x16 bf16) + C(16x16 f32)
    return __builtin_amdgcn_wmma_f32_16x16x32_bf16(
        /*neg_a=*/false, a, /*neg_b=*/false, b,
        /*c_mod=*/(short)0, c, /*reuse_a=*/false, /*reuse_b=*/false);
}

// Tensor Data Mover availability (this toolchain: 6-arg builtin, clang-23 /
// therock-10 form: (uint32x4, int32x8, int32x4, int32x4, int32x8, i32 cpol)).
#if defined(__AMDGCN__) && __has_builtin(__builtin_amdgcn_tensor_load_to_lds) && \
    __has_builtin(__builtin_amdgcn_s_wait_tensorcnt)
#define USE_TDM 1
#else
#define USE_TDM 0
#endif

#if USE_TDM
// Async DMA of a 64-row x 128-col bf16 tile (row stride 128 elems in memory)
// into LDS with +16B pad after each 256B row => 136-element LDS row stride.
__device__ __forceinline__ void tdm_load_tile_64x128(const bf16* gsrc, bf16* lds_dst) {
    unsigned lds_off = (unsigned)(unsigned long long)lds_dst;  // LDS aperture: addr[31:0]
    unsigned long long ga = (unsigned long long)gsrc;
    // D# group 0: count=1 | lds_addr | global_addr[56:0] | type=2 ("image")
    v4u g0 = { 1u,
               lds_off,
               (unsigned)ga,
               (unsigned)((ga >> 32) & 0x01FFFFFFull) | (2u << 30) };
    // D# group 1:
    //  w0: workgroup_mask=0, data_size=1 (2B), pad_enable=1,
    //      pad_interval=5 (64 DWORDs=256B), pad_amount=3 (4 DWORDs=16B)
    //  dims: tensor_dim0=128, tensor_dim1=64, tile_dim0=128, tile_dim1=64,
    //        tensor_dim0_stride=128
    v8i g1 = { (int)0x07510000,
               (int)(128u << 16),
               (int)(64u << 16),
               (int)(128u << 16),
               (int)64,
               (int)128,
               0, 0 };
    v4i gz4 = { 0, 0, 0, 0 };
    v8i gz8 = { 0, 0, 0, 0, 0, 0, 0, 0 };
    __builtin_amdgcn_tensor_load_to_lds(g0, g1, gz4, gz4, gz8, 0);
}
#endif

// ---------------------------------------------------------------------------
// f32 -> bf16 cast (vectorized x4)
// ---------------------------------------------------------------------------
__global__ __launch_bounds__(256) void cast_f32_bf16_kernel(
    const float* __restrict__ in, bf16* __restrict__ out, int n4) {
    int i = blockIdx.x * 256 + threadIdx.x;
    if (i < n4) {
        v4f f = *(const v4f*)(in + (size_t)i * 4);
        v4bf o;
        #pragma unroll
        for (int e = 0; e < 4; e++) o[e] = (bf16)f[e];
        *(v4bf*)(out + (size_t)i * 4) = o;
    }
}

// ---------------------------------------------------------------------------
// lambda = exp(sum(lq1*lk1)) - exp(sum(lq2*lk2)) + LAMBDA_INIT
// ---------------------------------------------------------------------------
__global__ void lambda_kernel(const float* __restrict__ lq1,
                              const float* __restrict__ lk1,
                              const float* __restrict__ lq2,
                              const float* __restrict__ lk2,
                              float* __restrict__ lam) {
    int l = threadIdx.x;  // 32 threads = 1 wave
    float s1 = 0.0f, s2 = 0.0f;
    for (int i = l; i < D_; i += 32) {
        s1 += lq1[i] * lk1[i];
        s2 += lq2[i] * lk2[i];
    }
    #pragma unroll
    for (int off = 16; off >= 1; off >>= 1) {
        s1 += __shfl_xor(s1, off, 32);
        s2 += __shfl_xor(s2, off, 32);
    }
    if (l == 0) {
        // LAMBDA_INIT = 0.8 - 0.6*exp(-0.3*16)
        *lam = __expf(s1) - __expf(s2) + 0.7950621517672336f;
    }
}

// ---------------------------------------------------------------------------
// GEMM: C[M,N] = A[M,K] * W[N,K]^T, bf16 in, f32 accumulate.
// Block tile 128x256, K-step 32, 8 waves; wave tile 64x64 (4x4 WMMA tiles)
// => 16 v_wmma per k-step against 16 ds_load_b128.
// heads_mode==1: write bf16 into [B,H,S,D] layout (projection path).
// heads_mode==0: write f32 row-major [M,N] (final output projection).
// ---------------------------------------------------------------------------
__global__ __launch_bounds__(256) void gemm_xwt_kernel(
    const bf16* __restrict__ A, const bf16* __restrict__ W,
    float* __restrict__ Cf, bf16* __restrict__ Ch,
    int M, int N, int K, int heads_mode) {
    __shared__ __align__(16) bf16 sA[128][40];  // +8 pad keeps 16B alignment
    __shared__ __align__(16) bf16 sB[256][40];

    const int tid  = threadIdx.x;
    const int lane = tid & 31;
    const int wave = tid >> 5;
    const int wm   = wave >> 2;  // 0..1
    const int wn   = wave & 3;   // 0..3
    const int m0   = blockIdx.x * 128;
    const int n0   = blockIdx.y * 256;
    const int mrow = lane & 15;
    const int koff = (lane >> 4) * 8;  // K chunk base per ISA 16-bit A layout

    v8f zero = {};
    v8f acc[4][4];
    #pragma unroll
    for (int i = 0; i < 4; i++)
        #pragma unroll
        for (int j = 0; j < 4; j++) acc[i][j] = zero;

    for (int kt = 0; kt < K; kt += 32) {
        // cooperative tile loads (chunks of 8 bf16 = 16B)
        for (int c = tid; c < 512; c += 256) {
            int r  = c >> 2;
            int ck = (c & 3) * 8;
            *(v8bf*)&sA[r][ck] = *(const v8bf*)(A + (size_t)(m0 + r) * K + kt + ck);
        }
        for (int c = tid; c < 1024; c += 256) {
            int r  = c >> 2;
            int ck = (c & 3) * 8;
            *(v8bf*)&sB[r][ck] = *(const v8bf*)(W + (size_t)(n0 + r) * K + kt + ck);
        }
        __syncthreads();

        FragBF af[4], bfr[4];
        #pragma unroll
        for (int i = 0; i < 4; i++) {
            int r = wm * 64 + i * 16 + mrow;
            af[i].h[0] = *(const v8bf*)&sA[r][koff];
            af[i].h[1] = *(const v8bf*)&sA[r][koff + 16];
        }
        #pragma unroll
        for (int j = 0; j < 4; j++) {
            int r = wn * 64 + j * 16 + mrow;
            bfr[j].h[0] = *(const v8bf*)&sB[r][koff];
            bfr[j].h[1] = *(const v8bf*)&sB[r][koff + 16];
        }
        #pragma unroll
        for (int i = 0; i < 4; i++)
            #pragma unroll
            for (int j = 0; j < 4; j++)
                acc[i][j] = wmma_bf16f32(af[i].v, bfr[j].v, acc[i][j]);
        __syncthreads();
    }

    const int rbase = (lane >> 4) * 8;  // C layout: M = r + 8*(lane/16)
    #pragma unroll
    for (int i = 0; i < 4; i++) {
        #pragma unroll
        for (int j = 0; j < 4; j++) {
            #pragma unroll
            for (int r = 0; r < 8; r++) {
                int m = m0 + wm * 64 + i * 16 + rbase + r;
                int n = n0 + wn * 64 + j * 16 + mrow;
                float val = acc[i][j][r];
                if (heads_mode) {
                    int b = m >> 11;          // m / S_
                    int s = m & (S_ - 1);
                    int h = n >> 7;           // n / D_
                    int d = n & (D_ - 1);
                    size_t off = (((size_t)(b * H_ + h)) * S_ + s) * D_ + d;
                    Ch[off] = (bf16)val;
                } else {
                    Cf[(size_t)m * N + n] = val;
                }
            }
        }
    }
}

// ---------------------------------------------------------------------------
// Flash-style two-stream differential attention.
// Grid (S/64, H, B); 128 threads = 4 waves; each wave owns 16 query rows.
// K1/K2 tiles are fetched by the Tensor Data Mover (async, TENSORcnt) when
// available; V is staged transposed by VALU so PV B-fragments are contiguous.
// ---------------------------------------------------------------------------
__device__ __forceinline__ void softmax_pv_stream(
    v8f s[4], float* __restrict__ mst, float* __restrict__ lst,
    v8f* __restrict__ o, bool diag, int qr0, int rbase, int key0,
    int mcol, int koff, float scale,
    bf16* __restrict__ sPw /* [16][72] */,
    const bf16* __restrict__ sVt /* [128][72] */) {
    // scale + causal mask (diagonal tile only)
    #pragma unroll
    for (int nt = 0; nt < 4; nt++) {
        #pragma unroll
        for (int r = 0; r < 8; r++) {
            float v = s[nt][r] * scale;
            if (diag) {
                int key = key0 + nt * 16 + mcol;
                int row = qr0 + rbase + r;
                if (key > row) v = -1e30f;
            }
            s[nt][r] = v;
        }
    }
    // online softmax update (rows live in 16-lane halves -> xor masks 8..1)
    float mnew[8], alpha[8];
    #pragma unroll
    for (int r = 0; r < 8; r++) {
        float mx = s[0][r];
        #pragma unroll
        for (int nt = 1; nt < 4; nt++) mx = fmaxf(mx, s[nt][r]);
        #pragma unroll
        for (int off = 8; off >= 1; off >>= 1) mx = fmaxf(mx, __shfl_xor(mx, off, 32));
        mnew[r]  = fmaxf(mst[r], mx);
        alpha[r] = __expf(mst[r] - mnew[r]);
        mst[r]   = mnew[r];
    }
    #pragma unroll
    for (int nt = 0; nt < 4; nt++)
        #pragma unroll
        for (int r = 0; r < 8; r++)
            s[nt][r] = __expf(s[nt][r] - mnew[r]);
    #pragma unroll
    for (int r = 0; r < 8; r++) {
        float sm = s[0][r] + s[1][r] + s[2][r] + s[3][r];
        #pragma unroll
        for (int off = 8; off >= 1; off >>= 1) sm += __shfl_xor(sm, off, 32);
        lst[r] = lst[r] * alpha[r] + sm;
    }
    #pragma unroll
    for (int t = 0; t < 8; t++)
        #pragma unroll
        for (int r = 0; r < 8; r++) o[t][r] *= alpha[r];

    // stage P (C-layout -> LDS) so we can reload it in A-fragment layout
    #pragma unroll
    for (int nt = 0; nt < 4; nt++)
        #pragma unroll
        for (int r = 0; r < 8; r++)
            sPw[(rbase + r) * 72 + nt * 16 + mcol] = (bf16)s[nt][r];

    // O += P(16x64) * V(64x128)  (V is staged transposed: sVt[d][key])
    FragBF pa[2];
    #pragma unroll
    for (int k2 = 0; k2 < 2; k2++) {
        pa[k2].h[0] = *(const v8bf*)(sPw + mcol * 72 + k2 * 32 + koff);
        pa[k2].h[1] = *(const v8bf*)(sPw + mcol * 72 + k2 * 32 + koff + 16);
    }
    #pragma unroll
    for (int t = 0; t < 8; t++) {
        #pragma unroll
        for (int k2 = 0; k2 < 2; k2++) {
            FragBF vb;
            const bf16* vp = sVt + (t * 16 + mcol) * 72 + k2 * 32;
            vb.h[0] = *(const v8bf*)(vp + koff);
            vb.h[1] = *(const v8bf*)(vp + koff + 16);
            o[t] = wmma_bf16f32(pa[k2].v, vb.v, o[t]);
        }
    }
}

__global__ __launch_bounds__(128) void diff_attn_kernel(
    const bf16* __restrict__ q1g, const bf16* __restrict__ q2g,
    const bf16* __restrict__ k1g, const bf16* __restrict__ k2g,
    const bf16* __restrict__ vg, const float* __restrict__ lam_p,
    bf16* __restrict__ outg) {
    __shared__ __align__(16) bf16 sK1[64][136];
    __shared__ __align__(16) bf16 sK2[64][136];
    __shared__ __align__(16) bf16 sVt[128][72];   // transposed V: [d][key]
    __shared__ __align__(16) bf16 sP[4][16][72];  // per-wave P staging

    const int tid   = threadIdx.x;
    const int lane  = tid & 31;
    const int wave  = tid >> 5;
    const int it    = blockIdx.x;  // query tile (64 rows)
    const int h     = blockIdx.y;
    const int b     = blockIdx.z;
    const size_t bh = ((size_t)(b * H_ + h)) * S_ * D_;
    const int qr0   = it * 64 + wave * 16;
    const int mcol  = lane & 15;
    const int koff  = (lane >> 4) * 8;
    const int rbase = (lane >> 4) * 8;
    const float scale = 0.08838834764831845f;  // 1/sqrt(128)
    const float lam   = *lam_p;

    // Q fragments straight from global (A layout: per-lane two 16B chunks)
    FragBF a1[4], a2[4];
    {
        const bf16* p1 = q1g + bh + (size_t)(qr0 + mcol) * D_;
        const bf16* p2 = q2g + bh + (size_t)(qr0 + mcol) * D_;
        #pragma unroll
        for (int dt = 0; dt < 4; dt++) {
            a1[dt].h[0] = *(const v8bf*)(p1 + dt * 32 + koff);
            a1[dt].h[1] = *(const v8bf*)(p1 + dt * 32 + koff + 16);
            a2[dt].h[0] = *(const v8bf*)(p2 + dt * 32 + koff);
            a2[dt].h[1] = *(const v8bf*)(p2 + dt * 32 + koff + 16);
        }
    }

    v8f zero = {};
    v8f o1[8], o2[8];
    float m1[8], l1[8], m2[8], l2[8];
    #pragma unroll
    for (int t = 0; t < 8; t++) { o1[t] = zero; o2[t] = zero; }
    #pragma unroll
    for (int r = 0; r < 8; r++) { m1[r] = -1e30f; l1[r] = 0.0f; m2[r] = -1e30f; l2[r] = 0.0f; }

    bf16* sPw = &sP[wave][0][0];

    for (int jt = 0; jt <= it; jt++) {
        const int key0 = jt * 64;
        const bf16* pk1 = k1g + bh + (size_t)key0 * D_;
        const bf16* pk2 = k2g + bh + (size_t)key0 * D_;
        const bf16* pv  = vg  + bh + (size_t)key0 * D_;
#if USE_TDM
        // TDM fetches K1/K2 tiles (padded rows) while all waves transpose V.
        if (wave == 0) {
            tdm_load_tile_64x128(pk1, &sK1[0][0]);
            tdm_load_tile_64x128(pk2, &sK2[0][0]);
            __builtin_amdgcn_s_wait_tensorcnt(0);
        }
        for (int c = tid; c < 1024; c += 128) {
            int r  = c >> 4;
            int ck = (c & 15) * 8;
            v8bf vv = *(const v8bf*)(pv + r * D_ + ck);
            #pragma unroll
            for (int e = 0; e < 8; e++) sVt[ck + e][r] = vv[e];
        }
#else
        for (int c = tid; c < 1024; c += 128) {
            int r  = c >> 4;
            int ck = (c & 15) * 8;
            *(v8bf*)&sK1[r][ck] = *(const v8bf*)(pk1 + r * D_ + ck);
            *(v8bf*)&sK2[r][ck] = *(const v8bf*)(pk2 + r * D_ + ck);
            v8bf vv = *(const v8bf*)(pv + r * D_ + ck);
            #pragma unroll
            for (int e = 0; e < 8; e++) sVt[ck + e][r] = vv[e];
        }
#endif
        __syncthreads();

        const bool diag = (jt == it);

        // ---- stream 1: S1 = Q1*K1^T ----
        {
            v8f s[4];
            #pragma unroll
            for (int nt = 0; nt < 4; nt++) s[nt] = zero;
            #pragma unroll
            for (int nt = 0; nt < 4; nt++) {
                const bf16* kb = &sK1[nt * 16 + mcol][0];
                #pragma unroll
                for (int dt = 0; dt < 4; dt++) {
                    FragBF bfr;
                    bfr.h[0] = *(const v8bf*)(kb + dt * 32 + koff);
                    bfr.h[1] = *(const v8bf*)(kb + dt * 32 + koff + 16);
                    s[nt] = wmma_bf16f32(a1[dt].v, bfr.v, s[nt]);
                }
            }
            softmax_pv_stream(s, m1, l1, o1, diag, qr0, rbase, key0, mcol, koff,
                              scale, sPw, &sVt[0][0]);
        }
        // ---- stream 2: S2 = Q2*K2^T ----
        {
            v8f s[4];
            #pragma unroll
            for (int nt = 0; nt < 4; nt++) s[nt] = zero;
            #pragma unroll
            for (int nt = 0; nt < 4; nt++) {
                const bf16* kb = &sK2[nt * 16 + mcol][0];
                #pragma unroll
                for (int dt = 0; dt < 4; dt++) {
                    FragBF bfr;
                    bfr.h[0] = *(const v8bf*)(kb + dt * 32 + koff);
                    bfr.h[1] = *(const v8bf*)(kb + dt * 32 + koff + 16);
                    s[nt] = wmma_bf16f32(a2[dt].v, bfr.v, s[nt]);
                }
            }
            softmax_pv_stream(s, m2, l2, o2, diag, qr0, rbase, key0, mcol, koff,
                              scale, sPw, &sVt[0][0]);
        }
        __syncthreads();
    }

    // epilogue: out = O1/l1 - lam * O2/l2, store bf16 [B,S,E]
    float inv1[8], inv2[8];
    #pragma unroll
    for (int r = 0; r < 8; r++) { inv1[r] = 1.0f / l1[r]; inv2[r] = 1.0f / l2[r]; }
    #pragma unroll
    for (int t = 0; t < 8; t++) {
        #pragma unroll
        for (int r = 0; r < 8; r++) {
            float val = o1[t][r] * inv1[r] - lam * (o2[t][r] * inv2[r]);
            int srow = qr0 + rbase + r;
            int col  = h * D_ + t * 16 + mcol;
            outg[((size_t)(b * S_ + srow)) * E_ + col] = (bf16)val;
        }
    }
}

// ---------------------------------------------------------------------------
// Host-side orchestration
// ---------------------------------------------------------------------------
extern "C" void kernel_launch(void* const* d_in, const int* in_sizes, int n_in,
                              void* d_out, int out_size, void* d_ws, size_t ws_size,
                              hipStream_t stream) {
    const float* x   = (const float*)d_in[0];
    const float* wq  = (const float*)d_in[1];
    const float* wk  = (const float*)d_in[2];
    const float* wv  = (const float*)d_in[3];
    const float* wq2 = (const float*)d_in[4];
    const float* wk2 = (const float*)d_in[5];
    const float* wo  = (const float*)d_in[6];
    const float* lq1 = (const float*)d_in[7];
    const float* lk1 = (const float*)d_in[8];
    const float* lq2 = (const float*)d_in[9];
    const float* lk2 = (const float*)d_in[10];
    float* out = (float*)d_out;

    const size_t XE = (size_t)B_ * S_ * E_;  // 8388608
    const size_t WE = (size_t)E_ * E_;       // 4194304

    char* ws = (char*)d_ws;
    float* lam = (float*)ws;
    bf16* xb   = (bf16*)(ws + 256);
    bf16* wqb  = xb + XE;
    bf16* wkb  = wqb + WE;
    bf16* wvb  = wkb + WE;
    bf16* wq2b = wvb + WE;
    bf16* wk2b = wq2b + WE;
    bf16* wob  = wk2b + WE;
    bf16* q1   = wob + WE;
    bf16* q2   = q1 + XE;
    bf16* k1   = q2 + XE;
    bf16* k2   = k1 + XE;
    bf16* v    = k2 + XE;
    bf16* attn = v + XE;

    // 1) casts to bf16
    cast_f32_bf16_kernel<<<(int)(XE / 4 / 256), 256, 0, stream>>>(x, xb, (int)(XE / 4));
    cast_f32_bf16_kernel<<<(int)(WE / 4 / 256), 256, 0, stream>>>(wq,  wqb,  (int)(WE / 4));
    cast_f32_bf16_kernel<<<(int)(WE / 4 / 256), 256, 0, stream>>>(wk,  wkb,  (int)(WE / 4));
    cast_f32_bf16_kernel<<<(int)(WE / 4 / 256), 256, 0, stream>>>(wv,  wvb,  (int)(WE / 4));
    cast_f32_bf16_kernel<<<(int)(WE / 4 / 256), 256, 0, stream>>>(wq2, wq2b, (int)(WE / 4));
    cast_f32_bf16_kernel<<<(int)(WE / 4 / 256), 256, 0, stream>>>(wk2, wk2b, (int)(WE / 4));
    cast_f32_bf16_kernel<<<(int)(WE / 4 / 256), 256, 0, stream>>>(wo,  wob,  (int)(WE / 4));

    // 2) lambda scalar
    lambda_kernel<<<1, 32, 0, stream>>>(lq1, lk1, lq2, lk2, lam);

    // 3) projections: [4096,2048] x [2048,2048]^T -> heads layout bf16
    const int M = B_ * S_, N = E_, K = E_;
    dim3 ggrid(M / 128, N / 256);  // (32,8)
    gemm_xwt_kernel<<<ggrid, 256, 0, stream>>>(xb, wqb,  nullptr, q1, M, N, K, 1);
    gemm_xwt_kernel<<<ggrid, 256, 0, stream>>>(xb, wq2b, nullptr, q2, M, N, K, 1);
    gemm_xwt_kernel<<<ggrid, 256, 0, stream>>>(xb, wkb,  nullptr, k1, M, N, K, 1);
    gemm_xwt_kernel<<<ggrid, 256, 0, stream>>>(xb, wk2b, nullptr, k2, M, N, K, 1);
    gemm_xwt_kernel<<<ggrid, 256, 0, stream>>>(xb, wvb,  nullptr, v,  M, N, K, 1);

    // 4) fused two-stream causal differential attention
    dim3 agrid(S_ / 64, H_, B_);  // (32,16,2)
    diff_attn_kernel<<<agrid, 128, 0, stream>>>(q1, q2, k1, k2, v, lam, attn);

    // 5) output projection -> f32 d_out
    gemm_xwt_kernel<<<ggrid, 256, 0, stream>>>(attn, wob, out, nullptr, M, N, K, 0);
}
